// xLSTMActorCritic_26637387170219
// MI455X (gfx1250) — matrix operand order repaired and checked by reading, not compiled, not run on hardware
//
#include <hip/hip_runtime.h>
#include <hip/hip_bf16.h>

// ---------------------------------------------------------------------------
// Types for CDNA5 WMMA
// ---------------------------------------------------------------------------
typedef __bf16 bf16_t;
typedef __attribute__((ext_vector_type(16))) __bf16 v16bf;
typedef __attribute__((ext_vector_type(8)))  __bf16 v8bf;
typedef __attribute__((ext_vector_type(8)))  float  v8f;

#define BD   128           // batch
#define SD   1024          // sequence
#define HD   256           // hidden
#define MS   (BD * SD)     // flattened rows = 131072

__device__ inline v8f wmma_bf16(v16bf a, v16bf b, v8f c) {
  return __builtin_amdgcn_wmma_f32_16x16x32_bf16(false, a, false, b, (short)0, c,
                                                 false, false);
}

// A fragment (16x32 bf16): lane r holds row M=r; per-lane ks are 16 contiguous
// elements at [half*16 .. half*16+15].
__device__ inline v16bf load_a_frag(const bf16_t* p /* row base at k0 */, int half) {
  v8bf c0 = *(const v8bf*)(p + half * 16);
  v8bf c1 = *(const v8bf*)(p + half * 16 + 8);
  v16bf a;
#pragma unroll
  for (int i = 0; i < 8; ++i) { a[i] = c0[i]; a[8 + i] = c1[i]; }
  return a;
}

// B fragment (32x16 bf16) where element (k,n) is contiguous in k for fixed n:
// lane holds 16 contiguous bf16 starting at (n-row base) + half*16.
__device__ inline v16bf load_b_contig(const bf16_t* p) {
  v8bf c0 = *(const v8bf*)(p);
  v8bf c1 = *(const v8bf*)(p + 8);
  v16bf b;
#pragma unroll
  for (int i = 0; i < 8; ++i) { b[i] = c0[i]; b[8 + i] = c1[i]; }
  return b;
}

__device__ inline float half_red_sum(float v) {
#pragma unroll
  for (int m = 1; m <= 8; m <<= 1) v += __shfl_xor(v, m, 32);
  return v;
}
__device__ inline float half_red_max(float v) {
#pragma unroll
  for (int m = 1; m <= 8; m <<= 1) v = fmaxf(v, __shfl_xor(v, m, 32));
  return v;
}
__device__ inline float wave_red_sum(float v) {
#pragma unroll
  for (int m = 1; m <= 16; m <<= 1) v += __shfl_xor(v, m, 32);
  return v;
}
__device__ inline float gelu_f(float x) {
  return 0.5f * x * (1.0f + erff(x * 0.70710678118654752f));
}
__device__ inline float clampf(float x, float lo, float hi) {
  return fminf(fmaxf(x, lo), hi);
}

// ---------------------------------------------------------------------------
// Software grid barrier (16 resident blocks, generation counter + s_sleep)
// ---------------------------------------------------------------------------
__device__ inline void grid_barrier(int* cnt, int* gen, int nblocks) {
  __syncthreads();
  if (threadIdx.x == 0) {
    __threadfence();
    int g = __hip_atomic_load(gen, __ATOMIC_ACQUIRE, __HIP_MEMORY_SCOPE_AGENT);
    int prev = __hip_atomic_fetch_add(cnt, 1, __ATOMIC_ACQ_REL, __HIP_MEMORY_SCOPE_AGENT);
    if (prev == nblocks - 1) {
      __hip_atomic_store(cnt, 0, __ATOMIC_RELAXED, __HIP_MEMORY_SCOPE_AGENT);
      __hip_atomic_fetch_add(gen, 1, __ATOMIC_ACQ_REL, __HIP_MEMORY_SCOPE_AGENT);
    } else {
      while (__hip_atomic_load(gen, __ATOMIC_ACQUIRE, __HIP_MEMORY_SCOPE_AGENT) == g) {
        __builtin_amdgcn_s_sleep(2);
      }
    }
  }
  __syncthreads();
}

// ---------------------------------------------------------------------------
// fp32 -> bf16 convert
// ---------------------------------------------------------------------------
__global__ void cvt_f32_bf16(const float* __restrict__ s, bf16_t* __restrict__ d,
                             long n) {
  long i = (long)blockIdx.x * blockDim.x + threadIdx.x;
  long stride = (long)gridDim.x * blockDim.x;
  for (; i < n; i += stride) d[i] = (bf16_t)s[i];
}

// ---------------------------------------------------------------------------
// Generic WMMA GEMM: C[M,N] = A[M,K](bf16,row-major) @ W[N,K]^T + bias
// block = 256 thr (8 waves), tile 128(M) x 64(N); W strip staged in LDS.
// ---------------------------------------------------------------------------
__global__ void __launch_bounds__(256)
gemm_bf16(const bf16_t* __restrict__ A, int lda,
          const bf16_t* __restrict__ W, int ldw,
          const float* __restrict__ bias,
          float* Cf, bf16_t* Cbf, const float* residF,
          int K, int ldc) {
  __shared__ bf16_t sW[64 * 520];
  const int tid = threadIdx.x;
  const int n0 = blockIdx.x * 64;
  const int stride = K + 8;
  {
    int row = tid >> 2, part = tid & 3, chunk = K >> 2;
    const bf16_t* src = W + (size_t)(n0 + row) * ldw + part * chunk;
    bf16_t* dst = sW + row * stride + part * chunk;
    for (int k = 0; k < chunk; k += 8) *(v8bf*)(dst + k) = *(const v8bf*)(src + k);
  }
  __syncthreads();
  const int wave = tid >> 5, lane = tid & 31, half = lane >> 4, r = lane & 15;
  const int m0 = blockIdx.y * 128 + wave * 16;
  v8f acc[4] = {};
  const int nk = K >> 5;
  const bf16_t* arow = A + (size_t)(m0 + r) * lda;
  for (int kb = 0; kb < nk; ++kb) {
    v16bf a = load_a_frag(arow + kb * 32, half);
#pragma unroll
    for (int t = 0; t < 4; ++t) {
      v16bf bfr = load_b_contig(sW + (t * 16 + r) * stride + kb * 32 + half * 16);
      acc[t] = wmma_bf16(a, bfr, acc[t]);
    }
  }
#pragma unroll
  for (int t = 0; t < 4; ++t) {
#pragma unroll
    for (int j = 0; j < 8; ++j) {
      int m = m0 + j + half * 8;
      int n = n0 + t * 16 + r;
      size_t off = (size_t)m * ldc + n;
      float v = acc[t][j] + bias[n];
      if (residF) v += residF[off];
      if (Cf)  Cf[off]  = v;
      if (Cbf) Cbf[off] = (bf16_t)v;
    }
  }
}

// ---------------------------------------------------------------------------
// Row LayerNorm over 256 cols; optional fp32 / bf16 outputs. 1 wave / row.
// ---------------------------------------------------------------------------
__global__ void __launch_bounds__(256)
ln_rows_256(const float* __restrict__ src, const float* __restrict__ g,
            const float* __restrict__ bsh, float* outF, bf16_t* outBF, int nrows) {
  int wave = threadIdx.x >> 5, lane = threadIdx.x & 31;
  int row = blockIdx.x * 8 + wave;
  if (row >= nrows) return;
  const float* p = src + (size_t)row * 256;
  float v[8], s = 0.f, s2 = 0.f;
#pragma unroll
  for (int i = 0; i < 8; ++i) {
    v[i] = p[lane + 32 * i];
    s += v[i]; s2 += v[i] * v[i];
  }
  s = wave_red_sum(s); s2 = wave_red_sum(s2);
  float mean = s * (1.f / 256.f);
  float var  = s2 * (1.f / 256.f) - mean * mean;
  float inv  = rsqrtf(var + 1e-5f);
#pragma unroll
  for (int i = 0; i < 8; ++i) {
    int c = lane + 32 * i;
    float o = (v[i] - mean) * inv * g[c] + bsh[c];
    if (outF)  outF[(size_t)row * 256 + c]  = o;
    if (outBF) outBF[(size_t)row * 256 + c] = (bf16_t)o;
  }
}

// ---------------------------------------------------------------------------
// Persistent sLSTM scan: 16 blocks; block bl owns hidden cols [bl*16, bl*16+16)
// gates = [xn_t, h_{t-1}] @ W^T + b, W[1024,512] strip (64 rows) held in LDS.
// h is double-buffered bf16 in global; c,n fp32. Grid barrier each step.
// Next step's xn slice is prefetched before the barrier to hide HBM latency
// behind the barrier wait (global_prefetch_b8).
// ---------------------------------------------------------------------------
__global__ void __launch_bounds__(256)
slstm_scan(const bf16_t* __restrict__ xn, const bf16_t* __restrict__ Wb,
           const float* __restrict__ bias, float* __restrict__ cur,
           bf16_t* __restrict__ hbuf, float* __restrict__ cst,
           float* __restrict__ nst, int* bar_cnt, int* bar_gen) {
  __shared__ bf16_t sW[64 * 520];
  const int tid = threadIdx.x, bl = blockIdx.x;
  const int j0 = bl * 16;
  {
    int row = tid >> 2, part = tid & 3;
    int wrow = (row >> 4) * 256 + j0 + (row & 15);   // gate g=row>>4, col j0+(row&15)
    const bf16_t* src = Wb + (size_t)wrow * 512 + part * 128;
    bf16_t* dst = sW + row * 520 + part * 128;
    for (int k = 0; k < 128; k += 8) *(v8bf*)(dst + k) = *(const v8bf*)(src + k);
  }
  for (int i = tid; i < BD * 16; i += 256) {  // init owned state slice
    int m = i >> 4, col = j0 + (i & 15);
    int idx = m * 256 + col;
    hbuf[idx] = (bf16_t)0.f; cst[idx] = 0.f; nst[idx] = 1.f;
  }
  __syncthreads();
  grid_barrier(bar_cnt, bar_gen, 16);

  const int wave = tid >> 5, lane = tid & 31, half = lane >> 4, r = lane & 15;
  const int m0 = wave * 16;
  const float bi = bias[j0 + r], bf = bias[256 + j0 + r],
              bo = bias[512 + j0 + r], bz = bias[768 + j0 + r];

  for (int t = 0; t < SD; ++t) {
    const bf16_t* hcur = hbuf + (t & 1) * (BD * HD);
    bf16_t* hnxt = hbuf + ((t + 1) & 1) * (BD * HD);
    v8f acc[4] = {};
    const bf16_t* xrow = xn + ((size_t)(m0 + r) * SD + t) * HD;
    const bf16_t* hrow = hcur + (m0 + r) * HD;
#pragma unroll
    for (int kb = 0; kb < 16; ++kb) {
      v16bf a = (kb < 8) ? load_a_frag(xrow + kb * 32, half)
                         : load_a_frag(hrow + (kb - 8) * 32, half);
#pragma unroll
      for (int g = 0; g < 4; ++g) {
        v16bf bfr = load_b_contig(sW + (g * 16 + r) * 520 + kb * 32 + half * 16);
        acc[g] = wmma_bf16(a, bfr, acc[g]);
      }
    }
    // Prefetch next timestep's xn slice (hidden behind the barrier wait).
    if (t + 1 < SD) {
      __builtin_prefetch(xrow + HD, 0, 1);        // row (m0+r), t+1, first half
      __builtin_prefetch(xrow + HD + 128, 0, 1);  // second cacheline half
    }
    const int col = j0 + r;
#pragma unroll
    for (int j = 0; j < 8; ++j) {
      int m = m0 + j + half * 8;
      float iv = __expf(clampf(acc[0][j] + bi, -5.f, 5.f));
      float fv = __expf(clampf(acc[1][j] + bf, -5.f, 5.f));
      float ov = acc[2][j] + bo;
      float zv = acc[3][j] + bz;
      int idx = m * 256 + col;
      float c2 = clampf(fv * cst[idx] + iv * tanhf(zv), -1e6f, 1e6f);
      float n2 = clampf(fv * nst[idx] + iv, 1e-6f, 1e6f);
      float h2 = (1.f / (1.f + __expf(-ov))) * (c2 / n2);
      if (!__builtin_isfinite(h2)) h2 = 0.f;
      cst[idx] = c2; nst[idx] = n2;
      hnxt[idx] = (bf16_t)h2;
      cur[((size_t)m * SD + t) * HD + col] += h2;   // residual
    }
    grid_barrier(bar_cnt, bar_gen, 16);
  }
}

// ---------------------------------------------------------------------------
// Flash attention: block = (q-tile of 128, head, batch); online softmax.
// qkv layout [B,S,768] bf16 (q|k|v). Out O [B,S,256] bf16 (pre-projection).
// ---------------------------------------------------------------------------
__global__ void __launch_bounds__(256)
flash_attn(const bf16_t* __restrict__ qkv, bf16_t* __restrict__ obuf) {
  __shared__ bf16_t sP[8 * 512];  // per-wave 16x32 P staging
  const int tid = threadIdx.x, wave = tid >> 5, lane = tid & 31;
  const int half = lane >> 4, r = lane & 15;
  const int qt = blockIdx.x, hh = blockIdx.y, b = blockIdx.z;
  const int m0 = qt * 128 + wave * 16;
  const bf16_t* base = qkv + (size_t)b * SD * 768;
  const bf16_t* qrow = base + (size_t)(m0 + r) * 768 + hh * 64;
  v16bf aq0 = load_a_frag(qrow, half);
  v16bf aq1 = load_a_frag(qrow + 32, half);
  v8f accO[4] = {};
  float mi[8], li[8];
#pragma unroll
  for (int j = 0; j < 8; ++j) { mi[j] = -3e38f; li[j] = 0.f; }
  bf16_t* sPw = sP + wave * 512;

  for (int kv0 = 0; kv0 < SD; kv0 += 32) {
    v8f s0 = {}, s1 = {};
    {
      const bf16_t* kp = base + (size_t)(kv0 + r) * 768 + 256 + hh * 64;
      s0 = wmma_bf16(aq0, load_b_contig(kp + half * 16), s0);
      s0 = wmma_bf16(aq1, load_b_contig(kp + 32 + half * 16), s0);
    }
    {
      const bf16_t* kp = base + (size_t)(kv0 + 16 + r) * 768 + 256 + hh * 64;
      s1 = wmma_bf16(aq0, load_b_contig(kp + half * 16), s1);
      s1 = wmma_bf16(aq1, load_b_contig(kp + 32 + half * 16), s1);
    }
    // Prefetch next K tile rows while we do softmax math.
    if (kv0 + 32 < SD) {
      __builtin_prefetch(base + (size_t)(kv0 + 32 + r) * 768 + 256 + hh * 64, 0, 1);
      __builtin_prefetch(base + (size_t)(kv0 + 48 + r) * 768 + 256 + hh * 64, 0, 1);
    }
    float p0[8], p1[8];
#pragma unroll
    for (int j = 0; j < 8; ++j) {
      float a0 = s0[j] * 0.125f, a1 = s1[j] * 0.125f;
      float mx = half_red_max(fmaxf(a0, a1));
      float mnew = fmaxf(mi[j], mx);
      float scale = __expf(mi[j] - mnew);
      float e0 = __expf(a0 - mnew), e1 = __expf(a1 - mnew);
      float rs = half_red_sum(e0 + e1);
      li[j] = li[j] * scale + rs;
      mi[j] = mnew;
      accO[0][j] *= scale; accO[1][j] *= scale;
      accO[2][j] *= scale; accO[3][j] *= scale;
      p0[j] = e0; p1[j] = e1;
    }
#pragma unroll
    for (int j = 0; j < 8; ++j) {   // stage P (C layout) -> A layout via LDS
      int ml = j + half * 8;
      sPw[ml * 32 + r]      = (bf16_t)p0[j];
      sPw[ml * 32 + 16 + r] = (bf16_t)p1[j];
    }
    v16bf ap = load_a_frag(sPw + r * 32, half);
#pragma unroll
    for (int t = 0; t < 4; ++t) {   // O += P @ V
      const bf16_t* vp = base + 512 + hh * 64 + t * 16 + r;
      v16bf bv;
#pragma unroll
      for (int j = 0; j < 8; ++j) {
        int k0 = half * 16 + 2 * j;
        bv[2 * j]     = vp[(size_t)(kv0 + k0) * 768];
        bv[2 * j + 1] = vp[(size_t)(kv0 + k0 + 1) * 768];
      }
      accO[t] = wmma_bf16(ap, bv, accO[t]);
    }
  }
#pragma unroll
  for (int t = 0; t < 4; ++t)
#pragma unroll
    for (int j = 0; j < 8; ++j) {
      float o = accO[t][j] / li[j];
      obuf[((size_t)b * SD + m0 + j + half * 8) * HD + hh * 64 + t * 16 + r] =
          (bf16_t)o;
    }
}

// ---------------------------------------------------------------------------
// Softmax attention pooling: block per batch row.
// ---------------------------------------------------------------------------
__global__ void __launch_bounds__(256)
pool_kernel(const float* __restrict__ xr, const float* __restrict__ pw,
            const float* __restrict__ pb, float* __restrict__ ctx) {
  __shared__ float sw[SD];
  __shared__ float red[256];
  const int b = blockIdx.x, tid = threadIdx.x;
  const float* xb = xr + (size_t)b * SD * HD;
  float lmax = -3e38f;
  for (int s = tid; s < SD; s += 256) {
    float acc = pb[0];
    const float* row = xb + (size_t)s * HD;
    for (int k = 0; k < HD; ++k) acc += row[k] * pw[k];
    sw[s] = acc; lmax = fmaxf(lmax, acc);
  }
  red[tid] = lmax; __syncthreads();
  for (int off = 128; off; off >>= 1) {
    if (tid < off) red[tid] = fmaxf(red[tid], red[tid + off]);
    __syncthreads();
  }
  float mx = red[0]; __syncthreads();
  float lsum = 0.f;
  for (int s = tid; s < SD; s += 256) {
    float e = __expf(sw[s] - mx); sw[s] = e; lsum += e;
  }
  red[tid] = lsum; __syncthreads();
  for (int off = 128; off; off >>= 1) {
    if (tid < off) red[tid] += red[tid + off];
    __syncthreads();
  }
  float inv = 1.f / red[0]; __syncthreads();
  float acc = 0.f;
  const int col = tid;
  for (int s = 0; s < SD; ++s) acc += sw[s] * xb[(size_t)s * HD + col];
  ctx[b * HD + col] = acc * inv;
}

// ---------------------------------------------------------------------------
// Tail (scalar fp32, negligible FLOPs)
// ---------------------------------------------------------------------------
__global__ void ie_kernel(const float* __restrict__ info, const float* w1,
                          const float* b1, const float* g, const float* bb,
                          const float* w2, const float* b2, float* out) {
  __shared__ float sv[64];
  __shared__ float red[64];
  const int b = blockIdx.x, t = threadIdx.x;  // 64 threads
  float acc = b1[t];
  for (int k = 0; k < 13; ++k) acc += info[b * 13 + k] * w1[t * 13 + k];
  red[t] = acc; __syncthreads();
  for (int off = 32; off; off >>= 1) { if (t < off) red[t] += red[t + off]; __syncthreads(); }
  float mean = red[0] / 64.f; __syncthreads();
  float d = acc - mean;
  red[t] = d * d; __syncthreads();
  for (int off = 32; off; off >>= 1) { if (t < off) red[t] += red[t + off]; __syncthreads(); }
  float var = red[0] / 64.f; __syncthreads();
  sv[t] = gelu_f(d * rsqrtf(var + 1e-5f) * g[t] + bb[t]);
  __syncthreads();
  float o = b2[t];
  for (int k = 0; k < 64; ++k) o += sv[k] * w2[t * 64 + k];
  out[b * 64 + t] = o;
}

__global__ void st1_kernel(const float* __restrict__ ctx, const float* __restrict__ ie,
                           const float* w, const float* bias, const float* g,
                           const float* bb, float* out) {
  __shared__ float red[256];
  const int b = blockIdx.x, t = threadIdx.x;  // 256 threads
  float acc = bias[t];
  const float* wr = w + t * 320;
  for (int k = 0; k < 256; ++k) acc += ctx[b * 256 + k] * wr[k];
  for (int k = 0; k < 64; ++k)  acc += ie[b * 64 + k] * wr[256 + k];
  red[t] = acc; __syncthreads();
  for (int off = 128; off; off >>= 1) { if (t < off) red[t] += red[t + off]; __syncthreads(); }
  float mean = red[0] / 256.f; __syncthreads();
  float d = acc - mean;
  red[t] = d * d; __syncthreads();
  for (int off = 128; off; off >>= 1) { if (t < off) red[t] += red[t + off]; __syncthreads(); }
  float var = red[0] / 256.f;
  out[b * 256 + t] = gelu_f(d * rsqrtf(var + 1e-5f) * g[t] + bb[t]);
}

__global__ void st2_kernel(const float* __restrict__ s1, const float* w,
                           const float* bias, const float* g, const float* bb,
                           float* out) {
  __shared__ float red[128];
  const int b = blockIdx.x, t = threadIdx.x;  // 128 threads
  float acc = bias[t];
  for (int k = 0; k < 256; ++k) acc += s1[b * 256 + k] * w[t * 256 + k];
  red[t] = acc; __syncthreads();
  for (int off = 64; off; off >>= 1) { if (t < off) red[t] += red[t + off]; __syncthreads(); }
  float mean = red[0] / 128.f; __syncthreads();
  float d = acc - mean;
  red[t] = d * d; __syncthreads();
  for (int off = 64; off; off >>= 1) { if (t < off) red[t] += red[t + off]; __syncthreads(); }
  float var = red[0] / 128.f;
  out[b * 128 + t] = gelu_f(d * rsqrtf(var + 1e-5f) * g[t] + bb[t]);
}

__global__ void head_kernel(const float* __restrict__ s2, const float* a1w,
                            const float* a1b, const float* a2w, const float* a2b,
                            const float* c1w, const float* c1b, const float* cng,
                            const float* cnb, const float* c2w, const float* c2b,
                            float* out) {
  __shared__ float sa[64], sc[32], scg[32], sl[3];
  const int b = blockIdx.x, t = threadIdx.x;  // 64 threads
  const float* s = s2 + b * 128;
  float av = a1b[t];
  for (int k = 0; k < 128; ++k) av += s[k] * a1w[t * 128 + k];
  sa[t] = gelu_f(av);
  if (t < 32) {
    float cv = c1b[t];
    for (int k = 0; k < 128; ++k) cv += s[k] * c1w[t * 128 + k];
    sc[t] = cv;
  }
  __syncthreads();
  if (t < 32) {
    float m = 0.f;
    for (int k = 0; k < 32; ++k) m += sc[k];
    m *= (1.f / 32.f);
    float v = 0.f;
    for (int k = 0; k < 32; ++k) { float d = sc[k] - m; v += d * d; }
    v *= (1.f / 32.f);
    scg[t] = gelu_f((sc[t] - m) * rsqrtf(v + 1e-5f) * cng[t] + cnb[t]);
  }
  __syncthreads();
  if (t < 3) {
    float lg = a2b[t];
    for (int k = 0; k < 64; ++k) lg += sa[k] * a2w[t * 64 + k];
    sl[t] = lg;
  }
  __syncthreads();
  if (t == 0) {
    float mx = fmaxf(sl[0], fmaxf(sl[1], sl[2]));
    float e0 = __expf(sl[0] - mx), e1 = __expf(sl[1] - mx), e2 = __expf(sl[2] - mx);
    float inv = 1.f / (e0 + e1 + e2);
    out[b * 3 + 0] = e0 * inv; out[b * 3 + 1] = e1 * inv; out[b * 3 + 2] = e2 * inv;
    float val = c2b[0];
    for (int k = 0; k < 32; ++k) val += scg[k] * c2w[k];
    out[BD * 3 + b] = val;
  }
}

// ---------------------------------------------------------------------------
// Host orchestration
// ---------------------------------------------------------------------------
extern "C" void kernel_launch(void* const* d_in, const int* in_sizes, int n_in,
                              void* d_out, int out_size, void* d_ws, size_t ws_size,
                              hipStream_t stream) {
  (void)n_in; (void)out_size; (void)ws_size;
  const float* P[64];
  for (int i = 0; i < n_in && i < 64; ++i) P[i] = (const float*)d_in[i];

  const float *x_, *info_, *inpW, *inpB, *innG, *innB;
  const float *l0g, *l0b, *l0W, *l0bias, *l1g, *l1b, *l1W, *l1bias;
  const float *qkvW, *qkvB, *aoW, *aoB, *anG, *anB, *poolW, *poolB;
  const float *ie1W, *ie1B, *ienG, *ienB, *ie2W, *ie2B;
  const float *st1W, *st1B, *stn1G, *stn1B, *st2W, *st2B, *stn2G, *stn2B;
  const float *a1W, *a1B, *a2W, *a2B, *c1W, *c1B, *cnG, *cnB, *c2W, *c2B;

  if (in_sizes[0] == BD * SD * 64) {  // insertion-order flattening (x first)
    x_ = P[0]; info_ = P[1]; inpW = P[2]; inpB = P[3]; innG = P[4]; innB = P[5];
    l0g = P[6]; l0b = P[7]; l0W = P[8]; l0bias = P[9];
    l1g = P[10]; l1b = P[11]; l1W = P[12]; l1bias = P[13];
    qkvW = P[14]; qkvB = P[15]; aoW = P[16]; aoB = P[17];
    anG = P[18]; anB = P[19]; poolW = P[20]; poolB = P[21];
    ie1W = P[22]; ie1B = P[23]; ienG = P[24]; ienB = P[25]; ie2W = P[26]; ie2B = P[27];
    st1W = P[28]; st1B = P[29]; stn1G = P[30]; stn1B = P[31];
    st2W = P[32]; st2B = P[33]; stn2G = P[34]; stn2B = P[35];
    a1W = P[36]; a1B = P[37]; a2W = P[38]; a2B = P[39];
    c1W = P[40]; c1B = P[41]; cnG = P[42]; cnB = P[43]; c2W = P[44]; c2B = P[45];
  } else {  // jax.tree_leaves sorted-key flattening (info first, x last)
    info_ = P[0];
    a1W = P[1]; a1B = P[2]; a2W = P[3]; a2B = P[4]; anB = P[5]; anG = P[6];
    aoW = P[7]; aoB = P[8];
    c1W = P[9]; c1B = P[10]; c2W = P[11]; c2B = P[12]; cnB = P[13]; cnG = P[14];
    ie1W = P[15]; ie1B = P[16]; ie2W = P[17]; ie2B = P[18]; ienB = P[19]; ienG = P[20];
    innB = P[21]; innG = P[22]; inpW = P[23]; inpB = P[24];
    l0W = P[25]; l0bias = P[26]; l0b = P[27]; l0g = P[28];
    l1W = P[29]; l1bias = P[30]; l1b = P[31]; l1g = P[32];
    poolW = P[33]; poolB = P[34]; qkvW = P[35]; qkvB = P[36];
    st1W = P[37]; st1B = P[38]; st2W = P[39]; st2B = P[40];
    stn1B = P[41]; stn1G = P[42]; stn2B = P[43]; stn2G = P[44];
    x_ = P[45];
  }

  // Workspace carve-up
  char* ws = (char*)d_ws;
  size_t off = 0;
  auto alloc = [&](size_t bytes) {
    size_t o = off; off += (bytes + 255) & ~(size_t)255; return o;
  };
  float*  CUR  = (float*) (ws + alloc((size_t)MS * HD * 4));     // 134 MB
  bf16_t* XNBF = (bf16_t*)(ws + alloc((size_t)MS * HD * 2));     //  67 MB
  bf16_t* QKV  = (bf16_t*)(ws + alloc((size_t)MS * 768 * 2));    // 201 MB (reused as f32 tmp)
  bf16_t* OBF  = (bf16_t*)(ws + alloc((size_t)MS * HD * 2));     //  67 MB
  bf16_t* XBF  = (bf16_t*)(ws + alloc((size_t)MS * 64 * 2));     //  17 MB
  bf16_t* WINP = (bf16_t*)(ws + alloc((size_t)256 * 64 * 2));
  bf16_t* WL0  = (bf16_t*)(ws + alloc((size_t)1024 * 512 * 2));
  bf16_t* WL1  = (bf16_t*)(ws + alloc((size_t)1024 * 512 * 2));
  bf16_t* WQKV = (bf16_t*)(ws + alloc((size_t)768 * 256 * 2));
  bf16_t* WAO  = (bf16_t*)(ws + alloc((size_t)256 * 256 * 2));
  bf16_t* HBUF = (bf16_t*)(ws + alloc((size_t)2 * BD * HD * 2));
  float*  CST  = (float*) (ws + alloc((size_t)BD * HD * 4));
  float*  NST  = (float*) (ws + alloc((size_t)BD * HD * 4));
  float*  CTX  = (float*) (ws + alloc((size_t)BD * HD * 4));
  float*  IEB  = (float*) (ws + alloc((size_t)BD * 64 * 4));
  float*  S1B  = (float*) (ws + alloc((size_t)BD * 256 * 4));
  float*  S2B  = (float*) (ws + alloc((size_t)BD * 128 * 4));
  int*    BAR  = (int*)   (ws + alloc(256));
  float*  TMPF = (float*)QKV;  // fp32 view of the QKV region (embed pre-LN, x+ao)

  hipMemsetAsync(BAR, 0, 256, stream);

  // 0. bf16 conversions
  cvt_f32_bf16<<<2048, 256, 0, stream>>>(x_,  XBF,  (long)MS * 64);
  cvt_f32_bf16<<<64,  256, 0, stream>>>(inpW, WINP, 256 * 64);
  cvt_f32_bf16<<<512, 256, 0, stream>>>(l0W,  WL0,  1024 * 512);
  cvt_f32_bf16<<<512, 256, 0, stream>>>(l1W,  WL1,  1024 * 512);
  cvt_f32_bf16<<<256, 256, 0, stream>>>(qkvW, WQKV, 768 * 256);
  cvt_f32_bf16<<<128, 256, 0, stream>>>(aoW,  WAO,  256 * 256);

  // 1. embed: tmp = x @ inp_W^T + b ; cur = LN(tmp)
  gemm_bf16<<<dim3(4, MS / 128), 256, 0, stream>>>(XBF, 64, WINP, 64, inpB,
                                                   TMPF, nullptr, nullptr, 64, HD);
  ln_rows_256<<<MS / 8, 256, 0, stream>>>(TMPF, innG, innB, CUR, nullptr, MS);

  // 2. sLSTM layers (xn precompute + persistent scan)
  ln_rows_256<<<MS / 8, 256, 0, stream>>>(CUR, l0g, l0b, nullptr, XNBF, MS);
  slstm_scan<<<16, 256, 0, stream>>>(XNBF, WL0, l0bias, CUR, HBUF, CST, NST,
                                     BAR, BAR + 1);
  ln_rows_256<<<MS / 8, 256, 0, stream>>>(CUR, l1g, l1b, nullptr, XNBF, MS);
  slstm_scan<<<16, 256, 0, stream>>>(XNBF, WL1, l1bias, CUR, HBUF, CST, NST,
                                     BAR, BAR + 1);

  // 3. attention: qkv GEMM -> flash -> out-proj (+residual) -> LN -> pool
  cvt_f32_bf16<<<4096, 256, 0, stream>>>(CUR, XNBF, (long)MS * HD);
  gemm_bf16<<<dim3(12, MS / 128), 256, 0, stream>>>(XNBF, HD, WQKV, HD, qkvB,
                                                    nullptr, QKV, nullptr, HD, 768);
  flash_attn<<<dim3(SD / 128, 4, BD), 256, 0, stream>>>(QKV, OBF);
  gemm_bf16<<<dim3(4, MS / 128), 256, 0, stream>>>(OBF, HD, WAO, HD, aoB,
                                                   TMPF, nullptr, CUR, HD, HD);
  ln_rows_256<<<MS / 8, 256, 0, stream>>>(TMPF, anG, anB, CUR, nullptr, MS);
  pool_kernel<<<BD, 256, 0, stream>>>(CUR, poolW, poolB, CTX);

  // 4. tail
  ie_kernel<<<BD, 64, 0, stream>>>(info_, ie1W, ie1B, ienG, ienB, ie2W, ie2B, IEB);
  st1_kernel<<<BD, 256, 0, stream>>>(CTX, IEB, st1W, st1B, stn1G, stn1B, S1B);
  st2_kernel<<<BD, 128, 0, stream>>>(S1B, st2W, st2B, stn2G, stn2B, S2B);
  head_kernel<<<BD, 64, 0, stream>>>(S2B, a1W, a1B, a2W, a2B, c1W, c1B, cnG, cnB,
                                     c2W, c2B, (float*)d_out);
}
